// Encoder_472446403262
// MI455X (gfx1250) — compile-verified, matrix operand order checked by reading, not run on hardware
//
#include <hip/hip_runtime.h>

// ---------------- problem constants (from reference) ----------------
constexpr int LNUM = 6;       // layers
constexpr int CH   = 512;     // channels
constexpr int FFC  = 2048;    // ffn channels
constexpr int NH   = 8;       // heads
constexpr int WIN  = 4;       // rel window
constexpr int TT   = 1024;    // time steps
constexpr int NB   = 4;       // batch
constexpr int DH   = CH / NH; // 64 head dim
constexpr float SCALE = 0.125f; // 1/sqrt(64)

typedef __attribute__((ext_vector_type(16))) _Float16 v16h;
typedef __attribute__((ext_vector_type(8)))  _Float16 v8h;
typedef __attribute__((ext_vector_type(8)))  float    v8f;
typedef __attribute__((ext_vector_type(4)))  int      v4i;

// ---------------- CDNA5 async copy to LDS (guarded) ----------------
#if defined(__has_builtin)
#if __has_builtin(__builtin_amdgcn_global_load_async_to_lds_b128) && \
    __has_builtin(__builtin_amdgcn_s_wait_asynccnt)
#define USE_ASYNC_LDS 1
#endif
#endif
#ifndef USE_ASYNC_LDS
#define USE_ASYNC_LDS 0
#endif

__device__ __forceinline__ void async_copy16(void* lds, const void* g) {
#if USE_ASYNC_LDS
  __builtin_amdgcn_global_load_async_to_lds_b128(
      (__attribute__((address_space(1))) v4i*)(g),
      (__attribute__((address_space(3))) v4i*)(lds), 0, 0);
#else
  *(uint4*)lds = *(const uint4*)g;
#endif
}
__device__ __forceinline__ void async_join() {
#if USE_ASYNC_LDS
  __builtin_amdgcn_s_wait_asynccnt(0);
#endif
}

// k-index map for 16-bit A/B fragments of v_wmma_*_16x16x32_* (ISA 7.12.2):
// lanes 0-15 cover K {0..7,16..23}, lanes 16-31 cover K {8..15,24..31}.
__device__ __forceinline__ int kmap(int e, int lhi) {
  return (e < 8 ? e : e + 8) + (lhi ? 8 : 0);
}

__device__ __forceinline__ v8f wmma_f16(v16h a, v16h b, v8f c) {
  return __builtin_amdgcn_wmma_f32_16x16x32_f16(false, a, false, b,
                                                (short)0, c, false, false);
}

// Load one 16x32 fragment whose 16 halfs per lane are k-contiguous in two
// 8-half chunks: two ds_load_b128 + register concat.  base = row 0 of the
// 16-row tile, PITCH = LDS row pitch in halfs (multiple of 8).
template<int PITCH>
__device__ __forceinline__ v16h load_frag(const _Float16* base, int l15, int lhi) {
  const _Float16* r = base + (size_t)l15 * PITCH + (lhi ? 8 : 0);
  v8h lo = *(const v8h*)(r);
  v8h hi = *(const v8h*)(r + 16);
  return __builtin_shufflevector(lo, hi, 0, 1, 2, 3, 4, 5, 6, 7,
                                 8, 9, 10, 11, 12, 13, 14, 15);
}

// ---------------------------------------------------------------
// GEMM / conv1d:  Y[b,m,t] = act( sum_{ci,dk} Wh[dk,m,ci] * Xh[b,ci,t+dk-pad] + bias[m] )
// Wh: f16 [TAPS][Cout][Cin] (pre-converted, tap-major).  Xh: f16 [B][Cin][T]
// (buffer has >=16-half pad front/back so haloed b128 staging never faults).
// grid: (T/128, Cout/64, B); block 256 = 8 waves (2M x 4N); wave = 32M x 32N.
// ---------------------------------------------------------------
template<int TAPS, bool RELU, bool OUTF16>
__global__ __launch_bounds__(256) void gemm_wmma(
    const _Float16* __restrict__ Wh, const _Float16* __restrict__ Xh,
    const float* __restrict__ bias, float* __restrict__ Yf,
    _Float16* __restrict__ Yh, int Cout, int Cin, int T)
{
  constexpr int MT = 64, NT = 128, KT = 32;
  constexpr int AP = KT + 8;     // sA pitch (40 halfs, 80B, 16B-aligned rows)
  constexpr int BP = NT + 16;    // sB pitch (144 halfs), 8-half halo each side
  __shared__ alignas(16) _Float16 sA[TAPS][MT][AP];
  __shared__ alignas(16) _Float16 sB[KT][BP];

  const int b  = blockIdx.z;
  const _Float16* Xb = Xh + (size_t)b * Cin * T;
  const int m0 = blockIdx.y * MT;
  const int n0 = blockIdx.x * NT;
  const int tid = threadIdx.x;
  const int lane = tid & 31, wave = tid >> 5;
  const int lhi = lane >> 4, l15 = lane & 15;
  const int wm = (wave >> 2) * 32;      // 0 / 32
  const int wn = (wave & 3) * 32;       // 0 / 32 / 64 / 96

  v8f acc00 = {}, acc01 = {}, acc10 = {}, acc11 = {};

  for (int k0 = 0; k0 < Cin; k0 += KT) {
    __syncthreads();   // previous tile fully consumed
    // ---- stage W tiles: TAPS * 64x32 halfs, b128 copies ----
    for (int i = tid; i < TAPS * 256; i += 256) {
      int dk = i >> 8, rem = i & 255;
      int r = rem >> 2, c = (rem & 3) * 8;
      async_copy16(&sA[dk][r][c],
                   &Wh[((size_t)dk * Cout + m0 + r) * Cin + k0 + c]);
    }
    // ---- stage X tile with halo: 32 rows x 144 halfs ----
    for (int i = tid; i < KT * (BP / 8); i += 256) {
      int kk = i / (BP / 8);
      int c  = (i % (BP / 8)) * 8;
      async_copy16(&sB[kk][c], Xb + (size_t)(k0 + kk) * T + (n0 - 8 + c));
    }
    if (k0 + KT < Cin)  // prefetch next X tile (global_prefetch_b8)
      __builtin_prefetch(Xb + (size_t)(k0 + KT + (tid >> 3)) * T + n0 + (tid & 7) * 16, 0, 1);
    async_join();
    __syncthreads();
    if (TAPS == 3) {   // zero the conv boundary columns (t=-1 / t=T)
      if (n0 == 0 && tid < KT)      sB[tid][7]      = (_Float16)0;
      if (n0 + NT == T && tid < KT) sB[tid][8 + NT] = (_Float16)0;
      __syncthreads();
    }

    #pragma unroll
    for (int dk = 0; dk < TAPS; ++dk) {
      v16h a0 = load_frag<AP>(&sA[dk][wm][0],      l15, lhi);
      v16h a1 = load_frag<AP>(&sA[dk][wm + 16][0], l15, lhi);
      v16h b0, b1;
      const int col = 8 + wn + l15 + dk - (TAPS >> 1);
      #pragma unroll
      for (int e = 0; e < 16; ++e) {
        int kk = kmap(e, lhi);
        b0[e] = sB[kk][col];
        b1[e] = sB[kk][col + 16];
      }
      acc00 = wmma_f16(a0, b0, acc00);
      acc01 = wmma_f16(a0, b1, acc01);
      acc10 = wmma_f16(a1, b0, acc10);
      acc11 = wmma_f16(a1, b1, acc11);
    }
  }

  // ---- epilogue ----
  #pragma unroll
  for (int mi = 0; mi < 2; ++mi) {
    v8f* am[2] = { &acc00, &acc10 };
    v8f* an[2] = { &acc01, &acc11 };
    #pragma unroll
    for (int r = 0; r < 8; ++r) {
      int m = m0 + wm + mi * 16 + r + (lhi ? 8 : 0);
      float bv = bias[m];
      float v0 = (*am[mi])[r] + bv;
      float v1 = (*an[mi])[r] + bv;
      if (RELU) { v0 = fmaxf(v0, 0.0f); v1 = fmaxf(v1, 0.0f); }
      size_t o0 = (size_t)m * T + (n0 + wn + l15);
      size_t o1 = o0 + 16;
      if (OUTF16) {
        _Float16* Yb = Yh + (size_t)b * Cout * T;
        Yb[o0] = (_Float16)v0; Yb[o1] = (_Float16)v1;
      } else {
        float* Yb = Yf + (size_t)b * Cout * T;
        Yb[o0] = v0; Yb[o1] = v1;
      }
    }
  }
}

// ---------------------------------------------------------------
// rq[b,h,t,r] = SCALE * sum_d q[b,h,d,t] * rel[r,d]   (r = 0..8)
// ---------------------------------------------------------------
__global__ __launch_bounds__(256) void rq_kernel(
    const _Float16* __restrict__ q, const float* __restrict__ rel,
    float* __restrict__ rq, int T)
{
  __shared__ float srel[9 * DH];
  for (int i = threadIdx.x; i < 9 * DH; i += 256) srel[i] = rel[i];
  __syncthreads();
  const int b = blockIdx.z, h = blockIdx.y;
  const int t = blockIdx.x * 256 + threadIdx.x;
  const _Float16* qh = q + ((size_t)b * NH + h) * DH * T;
  float acc[9] = {};
  for (int d = 0; d < DH; ++d) {
    float qd = (float)qh[(size_t)d * T + t] * SCALE;
    #pragma unroll
    for (int r = 0; r < 9; ++r) acc[r] += qd * srel[r * DH + d];
  }
  float* out = rq + (((size_t)b * NH + h) * T + t) * 9;
  #pragma unroll
  for (int r = 0; r < 9; ++r) out[r] = acc[r];
}

// ---------------------------------------------------------------
// Flash attention.  q,k,v: f16 [b,h,d,t].  scores = SCALE*q.k + clamped rel.
// grid: (T/64, H, B); block 128 = 4 waves; wave owns 16 queries.
// ---------------------------------------------------------------
__global__ __launch_bounds__(128) void attn_wmma(
    const _Float16* __restrict__ q, const _Float16* __restrict__ k,
    const _Float16* __restrict__ v, const float* __restrict__ rq,
    _Float16* __restrict__ att, int T)
{
  constexpr int SP = 40;  // sK/sV pitch
  __shared__ alignas(16) _Float16 sK[DH][SP];     // [d][s]
  __shared__ alignas(16) _Float16 sV[DH][SP];     // [d][s]
  __shared__ alignas(16) _Float16 sP[4][16][32];  // per-wave P re-layout

  const int b = blockIdx.z, h = blockIdx.y;
  const size_t bh = (size_t)b * NH + h;
  const _Float16* qh = q + bh * DH * T;
  const _Float16* kh = k + bh * DH * T;
  const _Float16* vh = v + bh * DH * T;
  const float* rqh = rq + bh * T * 9;
  const int tid = threadIdx.x, lane = tid & 31, wave = tid >> 5;
  const int lhi = lane >> 4, l15 = lane & 15;
  const int t0 = blockIdx.x * 64 + wave * 16;

  // Q as two A fragments (d 0..31 / 32..63), pre-scaled (x0.125 exact in f16)
  v16h aq0, aq1;
  #pragma unroll
  for (int e = 0; e < 16; ++e) {
    int d = kmap(e, lhi);
    aq0[e] = qh[(size_t)d * T + (t0 + l15)] * (_Float16)0.125f;
    aq1[e] = qh[(size_t)(d + 32) * T + (t0 + l15)] * (_Float16)0.125f;
  }

  float mrow[8], lrow[8];
  #pragma unroll
  for (int r = 0; r < 8; ++r) { mrow[r] = -1e30f; lrow[r] = 0.0f; }
  v8f o0 = {}, o1 = {}, o2 = {}, o3 = {};

  for (int s0 = 0; s0 < T; s0 += 32) {
    __syncthreads();
    // stage K/V tiles (64 d x 32 s halfs each): 512 b128 copies, 4/thread
    #pragma unroll
    for (int i = 0; i < 4; ++i) {
      int idx = i * 128 + tid;                 // 0..511
      int tensor = idx >> 8, d = (idx >> 2) & 63, c = (idx & 3) * 8;
      const _Float16* src = (tensor ? vh : kh) + (size_t)d * T + s0 + c;
      async_copy16(tensor ? &sV[d][c] : &sK[d][c], src);
    }
    async_join();
    __syncthreads();

    // scores: two 16x16 C frags (keys s0..15 / s0+16..31), 4 WMMAs
    v8f s_lo = {}, s_hi = {};
    {
      v16h b00, b10, b01, b11;
      #pragma unroll
      for (int e = 0; e < 16; ++e) {
        int kk = kmap(e, lhi);
        b00[e] = sK[kk][l15];
        b10[e] = sK[kk + 32][l15];
        b01[e] = sK[kk][l15 + 16];
        b11[e] = sK[kk + 32][l15 + 16];
      }
      s_lo = wmma_f16(aq0, b00, s_lo);
      s_lo = wmma_f16(aq1, b10, s_lo);
      s_hi = wmma_f16(aq0, b01, s_hi);
      s_hi = wmma_f16(aq1, b11, s_hi);
    }

    // rel term + online softmax
    #pragma unroll
    for (int r = 0; r < 8; ++r) {
      int tq = t0 + r + (lhi ? 8 : 0);
      const float* rr = rqh + (size_t)tq * 9;
      int j0 = s0 + l15, j1 = j0 + 16;
      int i0 = min(max(j0 - tq + WIN, 0), 2 * WIN);
      int i1 = min(max(j1 - tq + WIN, 0), 2 * WIN);
      float slo = s_lo[r] + rr[i0];
      float shi = s_hi[r] + rr[i1];

      float tmax = fmaxf(slo, shi);
      #pragma unroll
      for (int off = 8; off; off >>= 1) tmax = fmaxf(tmax, __shfl_xor(tmax, off, 16));
      float mnew = fmaxf(mrow[r], tmax);
      float corr = __expf(mrow[r] - mnew);
      float plo = __expf(slo - mnew);
      float phi = __expf(shi - mnew);
      float ps = plo + phi;
      #pragma unroll
      for (int off = 8; off; off >>= 1) ps += __shfl_xor(ps, off, 16);
      lrow[r] = lrow[r] * corr + ps;
      mrow[r] = mnew;
      o0[r] *= corr; o1[r] *= corr; o2[r] *= corr; o3[r] *= corr;
      sP[wave][r + (lhi ? 8 : 0)][l15]      = (_Float16)plo;
      sP[wave][r + (lhi ? 8 : 0)][l15 + 16] = (_Float16)phi;
    }

    // P x V -> O  (same-wave LDS RAW is in-order)
    v16h ap  = load_frag<32>(&sP[wave][0][0], l15, lhi);
    v16h bv0 = load_frag<SP>(&sV[0][0],  l15, lhi);
    v16h bv1 = load_frag<SP>(&sV[16][0], l15, lhi);
    v16h bv2 = load_frag<SP>(&sV[32][0], l15, lhi);
    v16h bv3 = load_frag<SP>(&sV[48][0], l15, lhi);
    o0 = wmma_f16(ap, bv0, o0);
    o1 = wmma_f16(ap, bv1, o1);
    o2 = wmma_f16(ap, bv2, o2);
    o3 = wmma_f16(ap, bv3, o3);
  }

  // normalize, scatter to f16 [b, h*DH+d, t]
  _Float16* ab = att + ((size_t)b * CH + (size_t)h * DH) * T;
  #pragma unroll
  for (int r = 0; r < 8; ++r) {
    float inv = 1.0f / lrow[r];
    int tq = t0 + r + (lhi ? 8 : 0);
    ab[(size_t)(l15)      * T + tq] = (_Float16)(o0[r] * inv);
    ab[(size_t)(l15 + 16) * T + tq] = (_Float16)(o1[r] * inv);
    ab[(size_t)(l15 + 32) * T + tq] = (_Float16)(o2[r] * inv);
    ab[(size_t)(l15 + 48) * T + tq] = (_Float16)(o3[r] * inv);
  }
}

// ---------------------------------------------------------------
// x = LN_channel(x + y); also emit f16 copy for downstream WMMA consumers.
// ---------------------------------------------------------------
__global__ __launch_bounds__(256) void add_ln_kernel(
    float* __restrict__ x, const float* __restrict__ y,
    const float* __restrict__ g, const float* __restrict__ be,
    _Float16* __restrict__ xh, int T)
{
  const int wave = threadIdx.x >> 5, lane = threadIdx.x & 31;
  const size_t bt = (size_t)blockIdx.x * 8 + wave;
  const int b = (int)(bt / T), t = (int)(bt % T);
  float* xb = x + (size_t)b * CH * T + t;
  const float* yb = y + (size_t)b * CH * T + t;
  _Float16* hb = xh + (size_t)b * CH * T + t;
  float vals[16];
  float s = 0.0f, s2 = 0.0f;
  #pragma unroll
  for (int i = 0; i < 16; ++i) {
    int c = lane * 16 + i;
    float vv = xb[(size_t)c * T] + yb[(size_t)c * T];
    vals[i] = vv; s += vv; s2 += vv * vv;
  }
  #pragma unroll
  for (int off = 16; off; off >>= 1) {
    s  += __shfl_xor(s, off, 32);
    s2 += __shfl_xor(s2, off, 32);
  }
  const float mean = s * (1.0f / CH);
  const float var  = s2 * (1.0f / CH) - mean * mean;
  const float rstd = rsqrtf(var + 1e-4f);
  #pragma unroll
  for (int i = 0; i < 16; ++i) {
    int c = lane * 16 + i;
    float o = (vals[i] - mean) * rstd * g[c] + be[c];
    xb[(size_t)c * T] = o;
    hb[(size_t)c * T] = (_Float16)o;
  }
}

// ---------------- weight / activation converters ----------------
__global__ __launch_bounds__(256) void cvt_f16_kernel(
    const float* __restrict__ in, _Float16* __restrict__ out, int n)
{
  int i = blockIdx.x * 256 + threadIdx.x;
  if (i < n) out[i] = (_Float16)in[i];
}

// in: [M][Cin][3]  ->  out: [3][M][Cin]   (mcin = M*Cin)
__global__ __launch_bounds__(256) void cvt_tap_kernel(
    const float* __restrict__ in, _Float16* __restrict__ out, int mcin)
{
  int i = blockIdx.x * 256 + threadIdx.x;
  if (i >= mcin) return;
  const float* s = in + (size_t)i * 3;
  out[i]                    = (_Float16)s[0];
  out[(size_t)mcin + i]     = (_Float16)s[1];
  out[(size_t)2 * mcin + i] = (_Float16)s[2];
}

// ---------------------------------------------------------------
extern "C" void kernel_launch(void* const* d_in, const int* in_sizes, int n_in,
                              void* d_out, int out_size, void* d_ws, size_t ws_size,
                              hipStream_t stream)
{
  const float* x_in = (const float*)d_in[0];
  const float* wq   = (const float*)d_in[1];
  const float* bq   = (const float*)d_in[2];
  const float* wk   = (const float*)d_in[3];
  const float* bk   = (const float*)d_in[4];
  const float* wv   = (const float*)d_in[5];
  const float* bv   = (const float*)d_in[6];
  const float* wo   = (const float*)d_in[7];
  const float* bo   = (const float*)d_in[8];
  const float* relk = (const float*)d_in[9];
  const float* g1   = (const float*)d_in[10];
  const float* be1  = (const float*)d_in[11];
  const float* w1   = (const float*)d_in[12];
  const float* bb1  = (const float*)d_in[13];
  const float* w2   = (const float*)d_in[14];
  const float* bb2  = (const float*)d_in[15];
  const float* g2   = (const float*)d_in[16];
  const float* be2  = (const float*)d_in[17];
  (void)in_sizes; (void)n_in; (void)out_size; (void)ws_size;

  const size_t N1 = (size_t)NB * CH * TT;

  char* p = (char*)d_ws;
  auto alloc_f32 = [&](size_t n) {
    float* r = (float*)p; p += ((n * 4 + 255) & ~(size_t)255); return r;
  };
  auto alloc_f16p = [&](size_t n) {   // padded: 16 halfs front/back
    _Float16* r = (_Float16*)p; p += (((n + 32) * 2 + 255) & ~(size_t)255);
    return r + 16;
  };

  float*    xcur = alloc_f32(N1);
  float*    yb   = alloc_f32(N1);
  float*    rqb  = alloc_f32((size_t)NB * NH * TT * 9);
  _Float16* xh   = alloc_f16p(N1);
  _Float16* qb   = alloc_f16p(N1);
  _Float16* kb   = alloc_f16p(N1);
  _Float16* vb   = alloc_f16p(N1);
  _Float16* atb  = alloc_f16p(N1);
  _Float16* h1h  = alloc_f16p((size_t)NB * FFC * TT);
  _Float16* wqh  = alloc_f16p((size_t)CH * CH);
  _Float16* wkh  = alloc_f16p((size_t)CH * CH);
  _Float16* wvh  = alloc_f16p((size_t)CH * CH);
  _Float16* woh  = alloc_f16p((size_t)CH * CH);
  _Float16* w1h  = alloc_f16p((size_t)3 * FFC * CH);
  _Float16* w2h  = alloc_f16p((size_t)3 * CH * FFC);

  (void)hipMemcpyAsync(xcur, x_in, N1 * sizeof(float), hipMemcpyDeviceToDevice, stream);
  cvt_f16_kernel<<<(int)(N1 / 256), 256, 0, stream>>>(x_in, xh, (int)N1);

  const dim3 gC(TT / 128, CH / 64, NB);
  const dim3 gF(TT / 128, FFC / 64, NB);
  const dim3 gA(TT / 64, NH, NB);
  const dim3 gR(TT / 256, NH, NB);
  const int  gL = NB * TT / 8;
  const int  nCC = CH * CH;

  for (int i = 0; i < LNUM; ++i) {
    cvt_f16_kernel<<<nCC / 256, 256, 0, stream>>>(wq + (size_t)i * nCC, wqh, nCC);
    cvt_f16_kernel<<<nCC / 256, 256, 0, stream>>>(wk + (size_t)i * nCC, wkh, nCC);
    cvt_f16_kernel<<<nCC / 256, 256, 0, stream>>>(wv + (size_t)i * nCC, wvh, nCC);
    cvt_f16_kernel<<<nCC / 256, 256, 0, stream>>>(wo + (size_t)i * nCC, woh, nCC);
    cvt_tap_kernel<<<FFC * CH / 256, 256, 0, stream>>>(w1 + (size_t)i * FFC * CH * 3, w1h, FFC * CH);
    cvt_tap_kernel<<<FFC * CH / 256, 256, 0, stream>>>(w2 + (size_t)i * CH * FFC * 3, w2h, CH * FFC);

    gemm_wmma<1, false, true><<<gC, 256, 0, stream>>>(wqh, xh, bq + i * CH, nullptr, qb, CH, CH, TT);
    gemm_wmma<1, false, true><<<gC, 256, 0, stream>>>(wkh, xh, bk + i * CH, nullptr, kb, CH, CH, TT);
    gemm_wmma<1, false, true><<<gC, 256, 0, stream>>>(wvh, xh, bv + i * CH, nullptr, vb, CH, CH, TT);
    rq_kernel<<<gR, 256, 0, stream>>>(qb, relk + (size_t)i * (2 * WIN + 1) * DH, rqb, TT);
    attn_wmma<<<gA, 128, 0, stream>>>(qb, kb, vb, rqb, atb, TT);
    gemm_wmma<1, false, false><<<gC, 256, 0, stream>>>(woh, atb, bo + i * CH, yb, nullptr, CH, CH, TT);
    add_ln_kernel<<<gL, 256, 0, stream>>>(xcur, yb, g1 + i * CH, be1 + i * CH, xh, TT);
    gemm_wmma<3, true, true><<<gF, 256, 0, stream>>>(w1h, xh, bb1 + i * FFC, nullptr, h1h, FFC, CH, TT);
    gemm_wmma<3, false, false><<<gC, 256, 0, stream>>>(w2h, h1h, bb2 + i * CH, yb, nullptr, CH, FFC, TT);
    add_ln_kernel<<<gL, 256, 0, stream>>>(xcur, yb, g2 + i * CH, be2 + i * CH, xh, TT);
  }

  (void)hipMemcpyAsync(d_out, xcur, N1 * sizeof(float), hipMemcpyDeviceToDevice, stream);
}
